// GRU_76819785056271
// MI455X (gfx1250) — compile-verified, matrix operand order checked by reading, not run on hardware
//
#include <hip/hip_runtime.h>
#include <hip/hip_bf16.h>

// GRU LM: S=128, B=64, V=10000, E=256, H=512, L=2
// bf16 weights/activations + V_WMMA_F32_16X16X32_BF16, fp32 hidden state and
// accumulators. Weights (15.7MB bf16) live in L2 (192MB) across all 128 steps.
// Register blocking: each wave32 computes the full M strip (4x 16x16 tiles)
// for one N tile, so the weight (B) fragment is loaded once and reused by 4
// WMMAs -> 4x less L2 weight traffic on the dominant logits GEMM.

#define S_ 128
#define B_ 64
#define V_ 10000
#define E_ 256
#define H_ 512

typedef __attribute__((ext_vector_type(16))) __bf16 v16bf;
typedef __attribute__((ext_vector_type(8)))  __bf16 v8bf;
typedef __attribute__((ext_vector_type(8)))  float  v8f;

// ---- WMMA helpers -----------------------------------------------------------

// Lane holds K = {sel*8..sel*8+7} U {16+sel*8..16+sel*8+7}; base = rowptr+sel*8
__device__ inline v16bf load_frag(const __bf16* base, int k0) {
    v8bf lo = *reinterpret_cast<const v8bf*>(base + k0);
    v8bf hi = *reinterpret_cast<const v8bf*>(base + k0 + 16);
    return __builtin_shufflevector(lo, hi, 0, 1, 2, 3, 4, 5, 6, 7,
                                           8, 9, 10, 11, 12, 13, 14, 15);
}

// Full-M column strip: D[0:64, n0:n0+16] += A[64,lda] x W[n0:n0+16, :]^T.
// B fragment loaded once per K-slab, reused by 4 WMMAs (M register blocking).
__device__ inline void wmma_col4(const __bf16* __restrict__ A, int lda,
                                 const __bf16* __restrict__ W, int ldw,
                                 int K, int n0, v8f c[4]) {
    const int lane = threadIdx.x & 31;
    const int row  = lane & 15;
    const int sel  = lane >> 4;
    const __bf16* pb  = W + (size_t)(n0 + row) * ldw + sel * 8;
    const __bf16* pa0 = A + (size_t)row * lda + sel * 8;
    const __bf16* pa1 = pa0 + (size_t)16 * lda;
    const __bf16* pa2 = pa0 + (size_t)32 * lda;
    const __bf16* pa3 = pa0 + (size_t)48 * lda;
    for (int k0 = 0; k0 < K; k0 += 32) {
        v16bf b  = load_frag(pb,  k0);
        v16bf a0 = load_frag(pa0, k0);
        v16bf a1 = load_frag(pa1, k0);
        v16bf a2 = load_frag(pa2, k0);
        v16bf a3 = load_frag(pa3, k0);
        c[0] = __builtin_amdgcn_wmma_f32_16x16x32_bf16(false, a0, false, b, (short)0, c[0], false, false);
        c[1] = __builtin_amdgcn_wmma_f32_16x16x32_bf16(false, a1, false, b, (short)0, c[1], false, false);
        c[2] = __builtin_amdgcn_wmma_f32_16x16x32_bf16(false, a2, false, b, (short)0, c[2], false, false);
        c[3] = __builtin_amdgcn_wmma_f32_16x16x32_bf16(false, a3, false, b, (short)0, c[3], false, false);
    }
}

__device__ inline float sigmoidf_(float x) { return 1.0f / (1.0f + expf(-x)); }

// ---- Kernels ----------------------------------------------------------------

__global__ void k_convert_bf16(const float* __restrict__ src,
                               __bf16* __restrict__ dst, int n) {
    int i = blockIdx.x * blockDim.x + threadIdx.x;
    if (i < n) dst[i] = (__bf16)src[i];
}

// hidden [2,B,H] fp32 -> h0f/h1f fp32 + bf16 images into comb h-parts
__global__ void k_init(const float* __restrict__ hidden,
                       float* __restrict__ h0f, float* __restrict__ h1f,
                       __bf16* __restrict__ comb0, __bf16* __restrict__ comb1) {
    int i = blockIdx.x * blockDim.x + threadIdx.x;   // 2*B*H threads
    int l = i / (B_ * H_);
    int rem = i % (B_ * H_);
    int b = rem / H_, n = rem % H_;
    float v = hidden[i];
    if (l == 0) {
        h0f[rem] = v;
        comb0[(size_t)b * (H_ + E_) + n] = (__bf16)v;
    } else {
        h1f[rem] = v;
        comb1[(size_t)b * (2 * H_) + n] = (__bf16)v;
    }
}

// embedding lookup for step t -> bf16 x-part of comb0 and combrh0
__global__ void k_embed(const int* __restrict__ inputs_t,
                        const float* __restrict__ emb_table,
                        __bf16* __restrict__ comb0,
                        __bf16* __restrict__ combrh0) {
    int i = blockIdx.x * blockDim.x + threadIdx.x;   // B*E threads
    int b = i / E_, e = i % E_;
    int idx = inputs_t[b];
    __bf16 v = (__bf16)emb_table[(size_t)idx * E_ + e];
    comb0[(size_t)b * (H_ + E_) + H_ + e]   = v;
    combrh0[(size_t)b * (H_ + E_) + H_ + e] = v;
}

// r & z gates: sigmoid(comb @ W^T + b). gate0 -> rh=h*r (bf16), gate1 -> z f32
__global__ void k_gates(const __bf16* __restrict__ A, int lda, int K,
                        const __bf16* __restrict__ Wr,
                        const __bf16* __restrict__ Wz,
                        const float* __restrict__ br,
                        const float* __restrict__ bz,
                        const float* __restrict__ hf,
                        __bf16* __restrict__ rhdst, int ldrh,
                        float* __restrict__ zbuf) {
    const int ntiles = H_ / 16;                       // 32
    int strip = blockIdx.x * (blockDim.x >> 5) + (threadIdx.x >> 5);
    if (strip >= 2 * ntiles) return;                  // 64 strips
    int gate = strip / ntiles;
    int n0   = (strip % ntiles) * 16;
    v8f c[4] = {};
    wmma_col4(A, lda, gate ? Wz : Wr, K, K, n0, c);
    int lane = threadIdx.x & 31;
    int col = n0 + (lane & 15);
    int sel = lane >> 4;
    float bv = (gate ? bz : br)[col];
    for (int mt = 0; mt < 4; ++mt)
        for (int r = 0; r < 8; ++r) {
            int m = mt * 16 + r + 8 * sel;
            float g = sigmoidf_(c[mt][r] + bv);
            if (gate == 0)
                rhdst[(size_t)m * ldrh + col] = (__bf16)(hf[m * H_ + col] * g);
            else
                zbuf[m * H_ + col] = g;
        }
}

// h_tilda = tanh(combrh @ Wh^T + b); h = (1-z)*h + z*h_tilda; fan out bf16 h
__global__ void k_htilde(const __bf16* __restrict__ A, int lda, int K,
                         const __bf16* __restrict__ Wh,
                         const float* __restrict__ bh,
                         const float* __restrict__ zbuf,
                         float* __restrict__ hf,
                         __bf16* __restrict__ d1, int ld1,
                         __bf16* __restrict__ d2, int ld2,
                         __bf16* __restrict__ d3, int ld3) {
    const int ntiles = H_ / 16;
    int strip = blockIdx.x * (blockDim.x >> 5) + (threadIdx.x >> 5);
    if (strip >= ntiles) return;                      // 32 strips
    int n0 = strip * 16;
    v8f c[4] = {};
    wmma_col4(A, lda, Wh, K, K, n0, c);
    int lane = threadIdx.x & 31;
    int col = n0 + (lane & 15);
    int sel = lane >> 4;
    float bv = bh[col];
    for (int mt = 0; mt < 4; ++mt)
        for (int r = 0; r < 8; ++r) {
            int m = mt * 16 + r + 8 * sel;
            float ht = tanhf(c[mt][r] + bv);
            float z  = zbuf[m * H_ + col];
            float hn = (1.0f - z) * hf[m * H_ + col] + z * ht;
            hf[m * H_ + col] = hn;
            __bf16 hb = (__bf16)hn;
            d1[(size_t)m * ld1 + col] = hb;
            if (d2) d2[(size_t)m * ld2 + col] = hb;
            if (d3) d3[(size_t)m * ld3 + col] = hb;
        }
}

// logits_t = h1 @ Wy^T + by : [64,512] x [512,10000]
__global__ void k_logits(const __bf16* __restrict__ A,
                         const __bf16* __restrict__ Wy,
                         const float* __restrict__ by,
                         float* __restrict__ out) {
    const int ntiles = V_ / 16;                       // 625
    int strip = blockIdx.x * (blockDim.x >> 5) + (threadIdx.x >> 5);
    if (strip >= ntiles) return;
    int n0 = strip * 16;
    v8f c[4] = {};
    wmma_col4(A, H_, Wy, H_, H_, n0, c);
    int lane = threadIdx.x & 31;
    int col = n0 + (lane & 15);
    int sel = lane >> 4;
    float bv = by[col];
    for (int mt = 0; mt < 4; ++mt)
        for (int r = 0; r < 8; ++r) {
            int m = mt * 16 + r + 8 * sel;
            out[(size_t)m * V_ + col] = c[mt][r] + bv;
        }
}

__global__ void k_final(const float* __restrict__ h0f,
                        const float* __restrict__ h1f,
                        float* __restrict__ out) {
    int i = blockIdx.x * blockDim.x + threadIdx.x;    // 2*B*H threads
    out[i] = (i < B_ * H_) ? h0f[i] : h1f[i - B_ * H_];
}

// ---- Host launch ------------------------------------------------------------

extern "C" void kernel_launch(void* const* d_in, const int* in_sizes, int n_in,
                              void* d_out, int out_size, void* d_ws, size_t ws_size,
                              hipStream_t stream) {
    (void)in_sizes; (void)n_in; (void)out_size; (void)ws_size;
    const int*   inputs = (const int*)  d_in[0];
    const float* hidden = (const float*)d_in[1];
    const float* embT   = (const float*)d_in[2];
    const float* Wr0 = (const float*)d_in[3];  const float* br0 = (const float*)d_in[4];
    const float* Wz0 = (const float*)d_in[5];  const float* bz0 = (const float*)d_in[6];
    const float* Wh0 = (const float*)d_in[7];  const float* bh0 = (const float*)d_in[8];
    const float* Wr1 = (const float*)d_in[9];  const float* br1 = (const float*)d_in[10];
    const float* Wz1 = (const float*)d_in[11]; const float* bz1 = (const float*)d_in[12];
    const float* Wh1 = (const float*)d_in[13]; const float* bh1 = (const float*)d_in[14];
    const float* Wy  = (const float*)d_in[15]; const float* by  = (const float*)d_in[16];
    float* out = (float*)d_out;

    // workspace carve-up (all offsets 256B aligned)
    char* p = (char*)d_ws;
    size_t off = 0;
    auto take = [&](size_t bytes) { char* q = p + off; off += (bytes + 255) & ~(size_t)255; return q; };
    __bf16* Wr0b = (__bf16*)take((size_t)H_ * (H_ + E_) * 2);
    __bf16* Wz0b = (__bf16*)take((size_t)H_ * (H_ + E_) * 2);
    __bf16* Wh0b = (__bf16*)take((size_t)H_ * (H_ + E_) * 2);
    __bf16* Wr1b = (__bf16*)take((size_t)H_ * (2 * H_) * 2);
    __bf16* Wz1b = (__bf16*)take((size_t)H_ * (2 * H_) * 2);
    __bf16* Wh1b = (__bf16*)take((size_t)H_ * (2 * H_) * 2);
    __bf16* Wyb  = (__bf16*)take((size_t)V_ * H_ * 2);
    __bf16* comb0   = (__bf16*)take((size_t)B_ * (H_ + E_) * 2);  // [h0 | emb]
    __bf16* combrh0 = (__bf16*)take((size_t)B_ * (H_ + E_) * 2);  // [h0*r0 | emb]
    __bf16* comb1   = (__bf16*)take((size_t)B_ * (2 * H_) * 2);   // [h1 | h0new]
    __bf16* combrh1 = (__bf16*)take((size_t)B_ * (2 * H_) * 2);   // [h1*r1 | h0new]
    float*  h0f  = (float*)take((size_t)B_ * H_ * 4);
    float*  h1f  = (float*)take((size_t)B_ * H_ * 4);
    float*  z0   = (float*)take((size_t)B_ * H_ * 4);
    float*  z1   = (float*)take((size_t)B_ * H_ * 4);
    __bf16* h1b  = (__bf16*)take((size_t)B_ * H_ * 2);

    auto cvt = [&](const float* s, __bf16* d, int n) {
        k_convert_bf16<<<(n + 255) / 256, 256, 0, stream>>>(s, d, n);
    };
    cvt(Wr0, Wr0b, H_ * (H_ + E_));
    cvt(Wz0, Wz0b, H_ * (H_ + E_));
    cvt(Wh0, Wh0b, H_ * (H_ + E_));
    cvt(Wr1, Wr1b, H_ * 2 * H_);
    cvt(Wz1, Wz1b, H_ * 2 * H_);
    cvt(Wh1, Wh1b, H_ * 2 * H_);
    cvt(Wy,  Wyb,  V_ * H_);

    k_init<<<(2 * B_ * H_) / 256, 256, 0, stream>>>(hidden, h0f, h1f, comb0, comb1);

    for (int t = 0; t < S_; ++t) {
        k_embed<<<(B_ * E_) / 256, 256, 0, stream>>>(
            inputs + (size_t)t * B_, embT, comb0, combrh0);

        // layer 0:  K = H+E = 768 ; 64 strips -> 8 blocks x 8 waves
        k_gates<<<8, 256, 0, stream>>>(comb0, H_ + E_, H_ + E_,
                                       Wr0b, Wz0b, br0, bz0, h0f,
                                       combrh0, H_ + E_, z0);
        k_htilde<<<4, 256, 0, stream>>>(combrh0, H_ + E_, H_ + E_, Wh0b, bh0,
                                        z0, h0f,
                                        comb0, H_ + E_,          // next-step h-part
                                        comb1 + H_, 2 * H_,      // L1 x-part
                                        combrh1 + H_, 2 * H_);   // L1 x-part (rh buf)

        // layer 1:  K = 2H = 1024
        k_gates<<<8, 256, 0, stream>>>(comb1, 2 * H_, 2 * H_,
                                       Wr1b, Wz1b, br1, bz1, h1f,
                                       combrh1, 2 * H_, z1);
        k_htilde<<<4, 256, 0, stream>>>(combrh1, 2 * H_, 2 * H_, Wh1b, bh1,
                                        z1, h1f,
                                        comb1, 2 * H_,           // next-step h-part
                                        h1b, H_,                 // logits A operand
                                        (__bf16*)nullptr, 0);

        // logits: 625 column strips, 8 waves/block -> 79 blocks
        k_logits<<<79, 256, 0, stream>>>(h1b, Wyb, by,
                                         out + (size_t)t * B_ * V_);
    }

    k_final<<<(2 * B_ * H_) / 256, 256, 0, stream>>>(h0f, h1f,
                                                     out + (size_t)S_ * B_ * V_);
}
